// GE2ELoss_70626442215524
// MI455X (gfx1250) — compile-verified, matrix operand order checked by reading, not run on hardware
//
#include <hip/hip_runtime.h>
#include <hip/hip_bf16.h>
#include <stdint.h>

#define N_SPK 2048
#define M_UTT 8
#define D_EMB 512
#define N_ROWS (N_SPK * M_UTT)   // 16384

typedef __attribute__((ext_vector_type(16))) __bf16 v16bf;
typedef __attribute__((ext_vector_type(8)))  float  v8f;

static __device__ __forceinline__ unsigned short f32_to_bf16_rne(float f) {
    unsigned int u = __float_as_uint(f);
    u += 0x7FFFu + ((u >> 16) & 1u);   // round-to-nearest-even
    return (unsigned short)(u >> 16);
}

// block-wide sum over 256 threads (8 wave32), result broadcast to all threads
static __device__ __forceinline__ float block_reduce_sum(float v, float* sm) {
    v += __shfl_xor(v, 1, 32);
    v += __shfl_xor(v, 2, 32);
    v += __shfl_xor(v, 4, 32);
    v += __shfl_xor(v, 8, 32);
    v += __shfl_xor(v, 16, 32);
    const int lane = threadIdx.x & 31;
    const int wv   = threadIdx.x >> 5;
    if (lane == 0) sm[wv] = v;
    __syncthreads();
    if (threadIdx.x == 0) {
        float s = 0.f;
        for (int i = 0; i < 8; ++i) s += sm[i];
        sm[8] = s;
    }
    __syncthreads();
    float r = sm[8];
    __syncthreads();
    return r;
}

// One block per speaker: centroid (bf16), leave-one-out diagonals, bf16 copy of e.
__global__ void ge2e_prep(const float* __restrict__ e,
                          unsigned short* __restrict__ e_bf,
                          unsigned short* __restrict__ c_bf,
                          float* __restrict__ diag) {
    __shared__ float sm[9];
    const int n   = blockIdx.x;
    const int tid = threadIdx.x;         // 256 threads, each owns dims tid, tid+256
    const int d0 = tid, d1 = tid + 256;

    float ev[M_UTT][2];
    float s0 = 0.f, s1 = 0.f;
    for (int m = 0; m < M_UTT; ++m) {
        const size_t r = (size_t)(n * M_UTT + m) * D_EMB;
        ev[m][0] = e[r + d0];
        ev[m][1] = e[r + d1];
        s0 += ev[m][0];
        s1 += ev[m][1];
        e_bf[r + d0] = f32_to_bf16_rne(ev[m][0]);
        e_bf[r + d1] = f32_to_bf16_rne(ev[m][1]);
    }

    // centroid = mean / ||mean|| = sum / ||sum||
    float ss  = block_reduce_sum(s0 * s0 + s1 * s1, sm);
    float inv = rsqrtf(ss);
    c_bf[(size_t)n * D_EMB + d0] = f32_to_bf16_rne(s0 * inv);
    c_bf[(size_t)n * D_EMB + d1] = f32_to_bf16_rne(s1 * inv);

    // diag[n,m] = dot(e_m, (sum - e_m)) / ||sum - e_m||   (the /(M-1) cancels)
    for (int m = 0; m < M_UTT; ++m) {
        float t0 = s0 - ev[m][0], t1 = s1 - ev[m][1];
        float dot = block_reduce_sum(ev[m][0] * t0 + ev[m][1] * t1, sm);
        float tt  = block_reduce_sum(t0 * t0 + t1 * t1, sm);
        if (tid == 0) diag[n * M_UTT + m] = dot * rsqrtf(tt);
    }
}

__global__ void ge2e_zero(float* out) { out[0] = 0.f; }

// 8 waves/block, 16 rows/wave => 128 rows/block. Fused bf16-WMMA GEMM + online softmax.
__global__ void __launch_bounds__(256)
ge2e_gemm_loss(const unsigned short* __restrict__ e_bf,
               const unsigned short* __restrict__ c_bf,
               const float* __restrict__ diag,
               const float* __restrict__ wp,
               const float* __restrict__ bp,
               float* __restrict__ out) {
    const int lane    = threadIdx.x & 31;
    const int wv      = threadIdx.x >> 5;
    const int half    = lane >> 4;      // K / row-half selector
    const int n16     = lane & 15;      // A: row within tile; B/D: column within tile
    const int rowbase = blockIdx.x * 128 + wv * 16;

    const float wabs = fabsf(wp[0]);
    const float bb   = bp[0];
    const float S    = wabs + bb;       // upper bound of logits (cosine sims <= 1)

    // Preload the wave's 16x512 bf16 A panel as 16 WMMA fragments (128 VGPRs).
    // 16-bit A 16x32 layout: lane m=L%16; half0 lanes: K=[0..7],[16..23]; half1: +8.
    uint4 areg[32];
    {
        const unsigned short* arow = e_bf + (size_t)(rowbase + n16) * D_EMB;
#pragma unroll
        for (int t = 0; t < 16; ++t) {
            areg[2 * t]     = *reinterpret_cast<const uint4*>(arow + t * 32 + 8 * half);
            areg[2 * t + 1] = *reinterpret_cast<const uint4*>(arow + t * 32 + 16 + 8 * half);
        }
    }

    // Per-lane softmax state: acc VGPR v holds row (rowbase + v + 8*half).
    float dl[8], runsum[8];
#pragma unroll
    for (int v = 0; v < 8; ++v) {
        dl[v]     = wabs * diag[rowbase + v + 8 * half] + bb;  // label logit (LOO centroid)
        runsum[v] = 0.f;
    }
    // label column = row/8; slot-invariant because rowbase is 16-aligned
    const int labelcol = (rowbase >> 3) + half;

    for (int ct = 0; ct < N_SPK / 16; ++ct) {
        const unsigned short* brow = c_bf + (size_t)(ct * 16 + n16) * D_EMB;
        v8f acc = {0.f, 0.f, 0.f, 0.f, 0.f, 0.f, 0.f, 0.f};
#pragma unroll
        for (int t = 0; t < 16; ++t) {
            union { uint4 q[2]; v16bf v; } a, b;
            a.q[0] = areg[2 * t];
            a.q[1] = areg[2 * t + 1];
            b.q[0] = *reinterpret_cast<const uint4*>(brow + t * 32 + 8 * half);
            b.q[1] = *reinterpret_cast<const uint4*>(brow + t * 32 + 16 + 8 * half);
            acc = __builtin_amdgcn_wmma_f32_16x16x32_bf16(
                /*neg_a=*/false, a.v, /*neg_b=*/false, b.v,
                /*c_mod=*/(short)0, acc, /*reuse_a=*/false, /*reuse_b=*/false);
        }
        const bool isd = (ct * 16 + n16) == labelcol;  // this lane owns the diagonal col
#pragma unroll
        for (int v = 0; v < 8; ++v) {
            float logit = isd ? dl[v] : (wabs * acc[v] + bb);
            runsum[v] += __expf(logit - S);
        }
    }

    // Reduce each row's partial sums across its 16 owner lanes; emit loss.
    float lossacc = 0.f;
#pragma unroll
    for (int v = 0; v < 8; ++v) {
        float tot = runsum[v];
        tot += __shfl_xor(tot, 1, 32);
        tot += __shfl_xor(tot, 2, 32);
        tot += __shfl_xor(tot, 4, 32);
        tot += __shfl_xor(tot, 8, 32);
        if (n16 == 0) lossacc += S + __logf(tot) - dl[v];   // logsumexp - label logit
    }
    if (n16 == 0) atomicAdd(out, lossacc * (1.0f / (float)N_ROWS));
}

extern "C" void kernel_launch(void* const* d_in, const int* in_sizes, int n_in,
                              void* d_out, int out_size, void* d_ws, size_t ws_size,
                              hipStream_t stream) {
    (void)in_sizes; (void)n_in; (void)out_size; (void)ws_size;
    const float* e = (const float*)d_in[0];   // [16384, 512] f32
    const float* w = (const float*)d_in[1];   // scalar
    const float* b = (const float*)d_in[2];   // scalar
    float* out = (float*)d_out;               // scalar loss

    char* ws = (char*)d_ws;
    unsigned short* e_bf = (unsigned short*)ws;                                   // 16 MB
    unsigned short* c_bf = (unsigned short*)(ws + (size_t)N_ROWS * D_EMB * 2);    //  2 MB
    float*          diag = (float*)(ws + (size_t)N_ROWS * D_EMB * 2
                                       + (size_t)N_SPK * D_EMB * 2);              // 64 KB

    ge2e_prep<<<N_SPK, 256, 0, stream>>>(e, e_bf, c_bf, diag);
    ge2e_zero<<<1, 1, 0, stream>>>(out);
    ge2e_gemm_loss<<<N_ROWS / 128, 256, 0, stream>>>(e_bf, c_bf, diag, w, b, out);
}
